// PWCNetWOX1SecondaryFlow_71648644431938
// MI455X (gfx1250) — matrix operand order repaired and checked by reading, not compiled, not run on hardware
//
#include <hip/hip_runtime.h>
#include <hip/hip_bf16.h>
#include <cstdint>
#include <cstddef>

// ---------------------------------------------------------------------------
// PWC-Net forward for MI455X (gfx1250).
// All 3x3 convolutions run as implicit GEMM on v_wmma_f32_16x16x32_bf16
// (bf16 A/B, f32 accumulate). Each wave computes a 16(M) x 16*NB(N) tile:
// the activation A-fragment is gathered once per K-step and reused across
// NB weight fragments (4 WMMAs in flight per K-step for the big convs).
// Activations live in bf16 NHWC with channel strides that are multiples of
// 8 so every A-fragment gather is two aligned 16-byte loads per lane.
// Weights are re-packed on device into the exact per-lane B-fragment layout
// each launch (deterministic). Working set (~150MB) fits in the 192MB L2.
// ---------------------------------------------------------------------------

typedef __attribute__((ext_vector_type(16))) __bf16         v16bf;
typedef __attribute__((ext_vector_type(8)))  float          v8f;
typedef __attribute__((ext_vector_type(16))) unsigned short v16u;
typedef __attribute__((ext_vector_type(8)))  unsigned short v8u;

static_assert(sizeof(v16bf) == 32, "v16bf must be 32 bytes");

struct Seg4 { int n; int relStart[4]; int refStart[4]; int len[4]; };

__device__ __forceinline__ unsigned short f2bf(float f) {
    unsigned int u = __builtin_bit_cast(unsigned int, f);
    u += 0x7FFFu + ((u >> 16) & 1u);           // round-to-nearest-even
    return (unsigned short)(u >> 16);
}
__device__ __forceinline__ float bf2f(unsigned short h) {
    unsigned int u = ((unsigned int)h) << 16;
    return __builtin_bit_cast(float, u);
}

// ---------------------------------------------------------------------------
// Zero fill (dwords)
// ---------------------------------------------------------------------------
__global__ void zero_kernel(unsigned int* p, long n) {
    long i = (long)blockIdx.x * blockDim.x + threadIdx.x;
    if (i < n) p[i] = 0u;
}

// ---------------------------------------------------------------------------
// NCHW f32 -> NHWC bf16 (channel-padded)
// ---------------------------------------------------------------------------
__global__ void to_nhwc_kernel(const float* __restrict__ src, unsigned short* __restrict__ dst,
                               int B, int C, int H, int W, int CT) {
    long idx = (long)blockIdx.x * blockDim.x + threadIdx.x;
    long total = (long)B * H * W;
    if (idx >= total) return;
    int x = (int)(idx % W);
    long t = idx / W;
    int y = (int)(t % H);
    int b = (int)(t / H);
    for (int c = 0; c < C; ++c)
        dst[idx * CT + c] = f2bf(src[((long)(b * C + c) * H + y) * W + x]);
}

// ---------------------------------------------------------------------------
// Pack OIHW f32 weights into per-lane WMMA B-fragment layout:
//   element index = (((coTile*k*k + tap)*chunks + cc)*32 + lane)*16 + slot
//   lane<16 : N = lane,    K(in chunk) = slot<8 ? slot : 8 + slot
//   lane>=16: N = lane-16, K(in chunk) = slot<8 ? 8 + slot : 16 + slot
// Segment map translates buffer-relative channel -> reference cin index
// (handles the corr/corr_s/flow concat layout). Out-of-range -> 0.
// ---------------------------------------------------------------------------
__global__ void pack_w_kernel(const float* __restrict__ w, unsigned short* __restrict__ out,
                              int cout, int cinRef, int k, int chunks, Seg4 segs) {
    long idx = (long)blockIdx.x * blockDim.x + threadIdx.x;
    int kk = k * k;
    int coTiles = (cout + 15) >> 4;
    long total = (long)coTiles * kk * chunks * 512;
    if (idx >= total) return;
    int s    = (int)(idx & 15);
    int lane = (int)((idx >> 4) & 31);
    long t2  = idx >> 9;
    int cc   = (int)(t2 % chunks); t2 /= chunks;
    int tap  = (int)(t2 % kk);
    int coT  = (int)(t2 / kk);
    int co   = coT * 16 + (lane & 15);
    int K    = (s < 8) ? (8 * (lane >> 4) + s) : (8 + 8 * (lane >> 4) + s);
    int ciRel = cc * 32 + K;
    float val = 0.f;
    if (co < cout) {
        for (int j = 0; j < segs.n; ++j) {
            int d = ciRel - segs.relStart[j];
            if (d >= 0 && d < segs.len[j]) {
                int ref = segs.refStart[j] + d;
                int ky = tap / k, kx = tap % k;
                val = w[(((long)co * cinRef + ref) * k + ky) * k + kx];
            }
        }
    }
    out[idx] = f2bf(val);
}

// ---------------------------------------------------------------------------
// Implicit-GEMM conv, one wave per 16(M) x 16*NB(N) output macro-tile.
//   A: 16 output pixels x 32 channels of one 3x3 tap (bf16 NHWC gather,
//      loaded ONCE per K-step, reused by NB WMMAs)
//   B: NB packed weight fragments (contiguous 32B load per lane each)
//   D = A*B + C via v_wmma_f32_16x16x32_bf16, f32 accumulate.
// Epilogue fuses bias + leaky ReLU, writes bf16 NHWC at channel outOff.
// All branches touching WMMA are wave-uniform (EXEC all-ones at WMMA).
// ---------------------------------------------------------------------------
template <int NB>
__global__ __launch_bounds__(256) void conv_wmma_kernel(
    const unsigned short* __restrict__ in, const unsigned short* __restrict__ wpk,
    const float* __restrict__ bias, unsigned short* __restrict__ out,
    int B, int Hin, int Win, int CinTot, int chOff,
    int Hout, int Wout, int Cout, int CoutTot, int outOff,
    int k, int stride, int dil, int chunks, int nGroups, int leaky) {

    int waveId = blockIdx.x * (blockDim.x >> 5) + (threadIdx.x >> 5);
    int lane   = threadIdx.x & 31;
    int M      = B * Hout * Wout;
    int mTiles = (M + 15) >> 4;
    if (waveId >= mTiles * nGroups) return;          // wave-uniform exit
    int mT = waveId / nGroups, g = waveId % nGroups;
    int tileM  = mT * 16;
    int coTiles = (Cout + 15) >> 4;
    int nAct   = coTiles - g * NB;                   // active sub-tiles (wave-uniform)
    if (nAct > NB) nAct = NB;

    int mRow  = tileM + (lane & 15);
    bool rowOk = (mRow < M);
    int mC = rowOk ? mRow : 0;
    int ox = mC % Wout;
    int tq = mC / Wout;
    int oy = tq % Hout;
    int bb = tq / Hout;
    int pad = (k - 1) * dil / 2;
    int hi  = lane >> 4;                              // K-half select per ISA layout
    int kk  = k * k;

    v8f acc[NB];
#pragma unroll
    for (int u = 0; u < NB; ++u) acc[u] = (v8f){};

    for (int tap = 0; tap < kk; ++tap) {
        int ky = tap / k, kx = tap % k;
        int iy = oy * stride - pad + ky * dil;
        int ix = ox * stride - pad + kx * dil;
        bool valid = rowOk && iy >= 0 && iy < Hin && ix >= 0 && ix < Win;
        long base = ((long)(bb * Hin + iy) * Win + ix) * CinTot + chOff;
        for (int cc = 0; cc < chunks; ++cc) {
            v16u au = {};
            if (valid) {
                const v8u* p0 = (const v8u*)(in + base + cc * 32 + (hi ? 8 : 0));
                const v8u* p1 = (const v8u*)(in + base + cc * 32 + (hi ? 24 : 16));
                v8u c0 = *p0, c1 = *p1;
#pragma unroll
                for (int i = 0; i < 8; ++i) { au[i] = c0[i]; au[i + 8] = c1[i]; }
            }
            v16bf a = __builtin_bit_cast(v16bf, au);
#pragma unroll
            for (int u = 0; u < NB; ++u) {
                if (u < nAct) {                        // wave-uniform guard
                    long nT = (long)g * NB + u;
                    v16u bu = *(const v16u*)(wpk + (((nT * kk + tap) * chunks + cc) * 32 + lane) * 16);
                    v16bf bfr = __builtin_bit_cast(v16bf, bu);
                    acc[u] = __builtin_amdgcn_wmma_f32_16x16x32_bf16(
                        false, a, false, bfr, (short)0, acc[u], false, false);
                }
            }
        }
    }
    // C/D layout: VGPR r -> M = tileM + r + 8*hi, N = tileN + (lane&15)
#pragma unroll
    for (int u = 0; u < NB; ++u) {
        if (u < nAct) {
            int n = (g * NB + u) * 16 + (lane & 15);
            float bv = (n < Cout) ? bias[n] : 0.f;
#pragma unroll
            for (int r = 0; r < 8; ++r) {
                int m = tileM + r + (hi ? 8 : 0);
                if (m < M && n < Cout) {
                    float v = acc[u][r] + bv;
                    if (leaky && v < 0.f) v *= 0.1f;
                    out[(long)m * CoutTot + outOff + n] = f2bf(v);
                }
            }
        }
    }
}

// ---------------------------------------------------------------------------
// Small scalar conv (cout==2 flow heads), f32 NCHW output, optional add.
// ---------------------------------------------------------------------------
__global__ void conv_scalar_kernel(const unsigned short* __restrict__ in,
                                   const float* __restrict__ w, const float* __restrict__ bias,
                                   float* __restrict__ out, const float* __restrict__ addSrc,
                                   int B, int H, int W, int CinTot, int chOff, int cinRef,
                                   Seg4 segs, int cout, int k, int dil) {
    long idx = (long)blockIdx.x * blockDim.x + threadIdx.x;
    long total = (long)B * cout * H * W;
    if (idx >= total) return;
    int x = (int)(idx % W);
    long t = idx / W;
    int y  = (int)(t % H); t /= H;
    int co = (int)(t % cout);
    int b  = (int)(t / cout);
    int pad = (k - 1) * dil / 2;
    float acc = bias[co];
    for (int ky = 0; ky < k; ++ky) {
        int iy = y - pad + ky * dil;
        if (iy < 0 || iy >= H) continue;
        for (int kx = 0; kx < k; ++kx) {
            int ix = x - pad + kx * dil;
            if (ix < 0 || ix >= W) continue;
            long pix = ((long)(b * H + iy) * W + ix) * CinTot + chOff;
            for (int j = 0; j < segs.n; ++j) {
                int rs = segs.relStart[j], fs = segs.refStart[j], ln = segs.len[j];
                for (int d = 0; d < ln; ++d) {
                    float a = bf2f(in[pix + rs + d]);
                    acc += a * w[(((long)co * cinRef + (fs + d)) * k + ky) * k + kx];
                }
            }
        }
    }
    if (addSrc) acc += addSrc[idx];
    out[idx] = acc;
}

// ---------------------------------------------------------------------------
// Correlation (mean over C) + leaky, md=4 -> 81 displacement channels.
// ---------------------------------------------------------------------------
__global__ void corr_kernel(const unsigned short* __restrict__ x1,
                            const unsigned short* __restrict__ x2,
                            unsigned short* __restrict__ xd,
                            int B, int C, int CT, int H, int W, int xdCT, int outOff) {
    long idx = (long)blockIdx.x * blockDim.x + threadIdx.x;
    long total = (long)B * 81 * H * W;
    if (idx >= total) return;
    int x = (int)(idx % W);
    long t = idx / W;
    int y = (int)(t % H); t /= H;
    int d = (int)(t % 81);
    int b = (int)(t / 81);
    int dy = d / 9 - 4, dx = d % 9 - 4;
    int y2 = y + dy, x2c = x + dx;
    float s = 0.f;
    if (y2 >= 0 && y2 < H && x2c >= 0 && x2c < W) {
        long p1 = ((long)(b * H + y) * W + x) * CT;
        long p2 = ((long)(b * H + y2) * W + x2c) * CT;
        for (int c = 0; c < C; ++c) s += bf2f(x1[p1 + c]) * bf2f(x2[p2 + c]);
    }
    float v = s / (float)C;
    if (v < 0.f) v *= 0.1f;
    xd[((long)(b * H + y) * W + x) * xdCT + outOff + d] = f2bf(v);
}

// ---------------------------------------------------------------------------
// Bilinear warp with validity mask (matches reference semantics).
// ---------------------------------------------------------------------------
__global__ void warp_kernel(const unsigned short* __restrict__ x, const float* __restrict__ flow,
                            unsigned short* __restrict__ outw,
                            int B, int C, int CT, int H, int W, int Him, int Wim) {
    long idx = (long)blockIdx.x * blockDim.x + threadIdx.x;
    long total = (long)B * H * W;
    if (idx >= total) return;
    int xq = (int)(idx % W);
    long t = idx / W;
    int yq = (int)(t % H);
    int b  = (int)(t / H);
    long hw = (long)H * W;
    float f0 = flow[((long)b * 2 + 0) * hw + (long)yq * W + xq];
    float f1 = flow[((long)b * 2 + 1) * hw + (long)yq * W + xq];
    float scx = (float)(W - 1) / ((float)((Wim - 1) > 1 ? (Wim - 1) : 1) * 0.05f);
    float scy = (float)(H - 1) / ((float)((Him - 1) > 1 ? (Him - 1) : 1) * 0.05f);
    float px = (float)xq + f0 * scx;
    float py = (float)yq + f1 * scy;
    float x0f = floorf(px), y0f = floorf(py);
    float fx = px - x0f, fy = py - y0f;
    int x0 = (int)x0f, y0 = (int)y0f;
    int xi[2] = { x0, x0 + 1 }, yi[2] = { y0, y0 + 1 };
    float m[2][2]; int xc[2], yc[2];
    for (int i = 0; i < 2; ++i) {
        xc[i] = xi[i] < 0 ? 0 : (xi[i] > W - 1 ? W - 1 : xi[i]);
        yc[i] = yi[i] < 0 ? 0 : (yi[i] > H - 1 ? H - 1 : yi[i]);
    }
    for (int j = 0; j < 2; ++j)
        for (int i = 0; i < 2; ++i)
            m[j][i] = (xi[i] >= 0 && xi[i] <= W - 1 && yi[j] >= 0 && yi[j] <= H - 1) ? 1.f : 0.f;
    float wgt[2][2] = { { (1 - fx) * (1 - fy), fx * (1 - fy) },
                        { (1 - fx) * fy,       fx * fy } };
    float maskv = wgt[0][0] * m[0][0] + wgt[0][1] * m[0][1] + wgt[1][0] * m[1][0] + wgt[1][1] * m[1][1];
    float mask = (maskv >= 0.9999f) ? 1.f : 0.f;
    long po = idx * CT;
    for (int c = 0; c < C; ++c) {
        float o = 0.f;
        for (int j = 0; j < 2; ++j)
            for (int i = 0; i < 2; ++i) {
                float v = m[j][i] * bf2f(x[((long)(b * H + yc[j]) * W + xc[i]) * CT + c]);
                o += v * wgt[j][i];
            }
        outw[po + c] = f2bf(o * mask);
    }
}

// ---------------------------------------------------------------------------
// Align-corners bilinear upsample (f32 NCHW), optional scale (final 1/DIV_FLOW)
// ---------------------------------------------------------------------------
__global__ void upsample_kernel(const float* __restrict__ in, float* __restrict__ out,
                                int B, int C, int Hi, int Wi, int Ho, int Wo, float scale) {
    long idx = (long)blockIdx.x * blockDim.x + threadIdx.x;
    long total = (long)B * C * Ho * Wo;
    if (idx >= total) return;
    int xo = (int)(idx % Wo);
    long t = idx / Wo;
    int yo = (int)(t % Ho); t /= Ho;
    int c  = (int)(t % C);
    int b  = (int)(t / C);
    float ys = (Ho > 1) ? (float)yo * (float)(Hi - 1) / (float)(Ho - 1) : 0.f;
    float xs = (Wo > 1) ? (float)xo * (float)(Wi - 1) / (float)(Wo - 1) : 0.f;
    int y0 = (int)floorf(ys), x0 = (int)floorf(xs);
    int y1 = y0 + 1 > Hi - 1 ? Hi - 1 : y0 + 1;
    int x1 = x0 + 1 > Wi - 1 ? Wi - 1 : x0 + 1;
    float fy = ys - (float)y0, fx = xs - (float)x0;
    long base = (long)(b * C + c) * Hi;
    float v00 = in[(base + y0) * Wi + x0], v01 = in[(base + y0) * Wi + x1];
    float v10 = in[(base + y1) * Wi + x0], v11 = in[(base + y1) * Wi + x1];
    float top = v00 * (1 - fx) + v01 * fx;
    float bot = v10 * (1 - fx) + v11 * fx;
    out[idx] = (top * (1 - fy) + bot * fy) * scale;
}

// ---------------------------------------------------------------------------
// Copy flow + flow_sec (f32 NCHW, 2ch each) into 4 bf16 channels of xd
// ---------------------------------------------------------------------------
__global__ void flow_to_xd_kernel(const float* __restrict__ fA, const float* __restrict__ fB,
                                  unsigned short* __restrict__ xd,
                                  int B, int H, int W, int CT, int off) {
    long idx = (long)blockIdx.x * blockDim.x + threadIdx.x;
    long hw = (long)H * W;
    long total = (long)B * hw;
    if (idx >= total) return;
    int b = (int)(idx / hw);
    long r = idx - (long)b * hw;
    xd[idx * CT + off + 0] = f2bf(fA[((long)b * 2 + 0) * hw + r]);
    xd[idx * CT + off + 1] = f2bf(fA[((long)b * 2 + 1) * hw + r]);
    xd[idx * CT + off + 2] = f2bf(fB[((long)b * 2 + 0) * hw + r]);
    xd[idx * CT + off + 3] = f2bf(fB[((long)b * 2 + 1) * hw + r]);
}

// ---------------------------------------------------------------------------
// Host orchestration
// ---------------------------------------------------------------------------
static inline int ceilDiv(int a, int b) { return (a + b - 1) / b; }

static Seg4 identSeg(int cin) {
    Seg4 s{}; s.n = 1; s.relStart[0] = 0; s.refStart[0] = 0; s.len[0] = cin; return s;
}
// xd layout (channels, stride 640):
//  [0,32) conv5 | [32,96) conv4 | [96,192) conv3 | [192,320) conv2 | [320,448) conv1
//  [448,529) corr | [536,617) corr_s | [624,628) flow,flow_sec | [628,632) new flows (ctx)
static Seg4 estSeg(int O, bool hasFlow, int flowLen) {
    Seg4 s{}; int nn = 0, r = 0;
    if (O < 448) { s.relStart[nn] = 0; s.refStart[nn] = 0; s.len[nn] = 448 - O; r = 448 - O; nn++; }
    s.relStart[nn] = 448 - O; s.refStart[nn] = r; s.len[nn] = 81; r += 81; nn++;
    s.relStart[nn] = 536 - O; s.refStart[nn] = r; s.len[nn] = 81; r += 81; nn++;
    if (hasFlow) { s.relStart[nn] = 624 - O; s.refStart[nn] = r; s.len[nn] = flowLen; r += flowLen; nn++; }
    s.n = nn; return s;
}

static void launch_pack(const float* w, unsigned short* dst, int cout, int cinRef,
                        int chunks, Seg4 sg, hipStream_t st) {
    long total = (long)ceilDiv(cout, 16) * 9 * chunks * 512;
    long blocks = (total + 255) / 256;
    pack_w_kernel<<<dim3((unsigned)blocks), dim3(256), 0, st>>>(w, dst, cout, cinRef, 3, chunks, sg);
}
static void launch_conv(const unsigned short* in, const unsigned short* wpk, const float* bias,
                        unsigned short* out, int B, int Hin, int Win, int CinTot, int chOff,
                        int Hout, int Wout, int Cout, int CoutTot, int outOff,
                        int stride, int dil, int chunks, int leaky, hipStream_t st) {
    int M = B * Hout * Wout;
    int mT = ceilDiv(M, 16);
    int coTiles = ceilDiv(Cout, 16);
    int NB = (coTiles >= 4) ? 4 : ((coTiles >= 2) ? 2 : 1);
    int nGroups = ceilDiv(coTiles, NB);
    long waves = (long)mT * nGroups;
    long blocks = (waves + 7) / 8;
    dim3 gd((unsigned)blocks), bd(256);
    if (NB == 4)
        conv_wmma_kernel<4><<<gd, bd, 0, st>>>(in, wpk, bias, out, B, Hin, Win, CinTot, chOff,
                                               Hout, Wout, Cout, CoutTot, outOff, 3, stride, dil,
                                               chunks, nGroups, leaky);
    else if (NB == 2)
        conv_wmma_kernel<2><<<gd, bd, 0, st>>>(in, wpk, bias, out, B, Hin, Win, CinTot, chOff,
                                               Hout, Wout, Cout, CoutTot, outOff, 3, stride, dil,
                                               chunks, nGroups, leaky);
    else
        conv_wmma_kernel<1><<<gd, bd, 0, st>>>(in, wpk, bias, out, B, Hin, Win, CinTot, chOff,
                                               Hout, Wout, Cout, CoutTot, outOff, 3, stride, dil,
                                               chunks, nGroups, leaky);
}

extern "C" void kernel_launch(void* const* d_in, const int* in_sizes, int n_in,
                              void* d_out, int out_size, void* d_ws, size_t ws_size,
                              hipStream_t stream) {
    (void)in_sizes; (void)n_in; (void)out_size; (void)ws_size;
    const int B = 2, Him = 384, Wim = 768;
    const int FC[7]  = { 3, 16, 32, 64, 96, 128, 196 };
    const int FCT[6] = { 32, 32, 64, 96, 128, 224 };
    const int Hs[6]  = { 192, 96, 48, 24, 12, 6 };
    const int Ws_[6] = { 384, 192, 96, 48, 24, 12 };

    // ---- parameter pointers (setup_inputs dict order) ----
    int pi = 2;
    const float *fpe_w[6][2], *fpe_b[6][2];
    for (int l = 0; l < 6; ++l)
        for (int j = 0; j < 2; ++j) { fpe_w[l][j] = (const float*)d_in[pi++]; fpe_b[l][j] = (const float*)d_in[pi++]; }
    const float *est_w[5][7], *est_b[5][7];
    for (int l = 0; l < 5; ++l)
        for (int j = 0; j < 7; ++j) { est_w[l][j] = (const float*)d_in[pi++]; est_b[l][j] = (const float*)d_in[pi++]; }
    const float *ctx_w[7], *ctx_b[7];
    for (int j = 0; j < 7; ++j) { ctx_w[j] = (const float*)d_in[pi++]; ctx_b[j] = (const float*)d_in[pi++]; }

    // ---- workspace layout ----
    uint8_t* W8 = (uint8_t*)d_ws;
    size_t cur = 0;
    auto alloc = [&](size_t bytes) -> void* {
        void* p = W8 + cur;
        cur = (cur + bytes + 255) & ~(size_t)255;
        return p;
    };
    unsigned short* inN[2];
    for (int i = 0; i < 2; ++i) inN[i] = (unsigned short*)alloc((size_t)B * Him * Wim * 8 * 2 + 4096);
    unsigned short* feat[2][6];
    for (int i = 0; i < 2; ++i)
        for (int s = 0; s < 6; ++s)
            feat[i][s] = (unsigned short*)alloc((size_t)B * Hs[s] * Ws_[s] * FCT[s] * 2 + 4096);
    unsigned short* tmpc1 = (unsigned short*)alloc((size_t)B * 192 * 384 * 32 * 2 + 4096);
    unsigned short* xd    = (unsigned short*)alloc((size_t)B * 96 * 192 * 640 * 2 + 4096);
    unsigned short* x2w   = (unsigned short*)alloc((size_t)4 << 20);
    unsigned short* x2ws  = (unsigned short*)alloc((size_t)4 << 20);
    unsigned short* cxA   = (unsigned short*)alloc((size_t)B * 96 * 192 * 128 * 2 + 4096);
    unsigned short* cxB   = (unsigned short*)alloc((size_t)B * 96 * 192 * 128 * 2 + 4096);
    float* fA = (float*)alloc((size_t)B * 2 * 96 * 192 * 4);
    float* fB = (float*)alloc((size_t)B * 2 * 96 * 192 * 4);
    float* tA = (float*)alloc((size_t)B * 2 * 96 * 192 * 4);
    float* tB = (float*)alloc((size_t)B * 2 * 96 * 192 * 4);
    size_t actEnd = cur;   // packed weights follow; fully written by pack kernels

    // zero activations (padding channels + initial flows must be 0)
    {
        long n = (long)(actEnd / 4);
        long blocks = (n + 255) / 256;
        zero_kernel<<<dim3((unsigned)blocks), dim3(256), 0, stream>>>((unsigned int*)d_ws, n);
    }

    auto grid1d = [](long n) { return dim3((unsigned)((n + 255) / 256)); };
    auto packBytes = [&](int cout, int chunks) -> size_t {
        return (size_t)ceilDiv(cout, 16) * 9 * chunks * 512 * 2;
    };

    // ---- feature pyramid extractor ----
    for (int img = 0; img < 2; ++img) {
        to_nhwc_kernel<<<grid1d((long)B * Him * Wim), dim3(256), 0, stream>>>(
            (const float*)d_in[img], inN[img], B, 3, Him, Wim, 8);
        const unsigned short* prev = inN[img];
        int pC = 3, pCT = 8, pH = Him, pW = Wim;
        for (int s = 0; s < 6; ++s) {
            int C = FC[s + 1], CT = FCT[s], H = Hs[s], W = Ws_[s];
            int ch1 = ceilDiv(pC, 32);
            unsigned short* wp1 = (unsigned short*)alloc(packBytes(C, ch1));
            launch_pack(fpe_w[s][0], wp1, C, pC, ch1, identSeg(pC), stream);
            launch_conv(prev, wp1, fpe_b[s][0], tmpc1, B, pH, pW, pCT, 0, H, W, C, CT, 0, 2, 1, ch1, 1, stream);
            int ch2 = ceilDiv(C, 32);
            unsigned short* wp2 = (unsigned short*)alloc(packBytes(C, ch2));
            launch_pack(fpe_w[s][1], wp2, C, C, ch2, identSeg(C), stream);
            launch_conv(tmpc1, wp2, fpe_b[s][1], feat[img][s], B, H, W, CT, 0, H, W, C, CT, 0, 1, 1, ch2, 1, stream);
            prev = feat[img][s]; pC = C; pCT = CT; pH = H; pW = W;
        }
    }

    // ---- coarse-to-fine flow estimation ----
    int pHf = 0, pWf = 0;
    for (int l = 0; l <= 4; ++l) {
        int s = 5 - l;
        int H = Hs[s], W = Ws_[s], C = FC[s + 1], CT = FCT[s];
        const unsigned short* x1 = feat[0][s];
        const unsigned short* x2 = feat[1][s];
        const unsigned short *xw, *xws;
        if (l == 0) {
            xw = x2; xws = x2;
        } else {
            upsample_kernel<<<grid1d((long)B * 2 * H * W), dim3(256), 0, stream>>>(fA, tA, B, 2, pHf, pWf, H, W, 1.f);
            { float* t = fA; fA = tA; tA = t; }
            upsample_kernel<<<grid1d((long)B * 2 * H * W), dim3(256), 0, stream>>>(fB, tB, B, 2, pHf, pWf, H, W, 1.f);
            { float* t = fB; fB = tB; tB = t; }
            warp_kernel<<<grid1d((long)B * H * W), dim3(256), 0, stream>>>(x2, fA, x2w, B, C, CT, H, W, Him, Wim);
            warp_kernel<<<grid1d((long)B * H * W), dim3(256), 0, stream>>>(x2, fB, x2ws, B, C, CT, H, W, Him, Wim);
            xw = x2w; xws = x2ws;
        }
        corr_kernel<<<grid1d((long)B * 81 * H * W), dim3(256), 0, stream>>>(x1, xw, xd, B, C, CT, H, W, 640, 448);
        corr_kernel<<<grid1d((long)B * 81 * H * W), dim3(256), 0, stream>>>(x1, xws, xd, B, C, CT, H, W, 640, 536);
        if (l > 0)
            flow_to_xd_kernel<<<grid1d((long)B * H * W), dim3(256), 0, stream>>>(fA, fB, xd, B, H, W, 640, 624);

        int chin = (l == 0) ? 162 : 166;
        struct EC { int O, cout, outOff; } ec[5] = {
            { 448, 128, 320 }, { 320, 128, 192 }, { 192, 96, 96 }, { 96, 64, 32 }, { 32, 32, 0 } };
        int top = (l > 0) ? 628 : 617;
        for (int j = 0; j < 5; ++j) {
            Seg4 sg = estSeg(ec[j].O, l > 0, 4);
            int chunks = ceilDiv(top - ec[j].O, 32);
            int cinRef = chin + (448 - ec[j].O);
            unsigned short* wp = (unsigned short*)alloc(packBytes(ec[j].cout, chunks));
            launch_pack(est_w[l][j], wp, ec[j].cout, cinRef, chunks, sg, stream);
            launch_conv(xd, wp, est_b[l][j], xd, B, H, W, 640, ec[j].O,
                        H, W, ec[j].cout, 640, ec[j].outOff, 1, 1, chunks, 1, stream);
        }
        // flow heads (cout=2) -> f32 NCHW
        Seg4 sgL = estSeg(0, l > 0, 4);
        conv_scalar_kernel<<<grid1d((long)B * 2 * H * W), dim3(256), 0, stream>>>(
            xd, est_w[l][5], est_b[l][5], fA, nullptr, B, H, W, 640, 0, chin + 448, sgL, 2, 3, 1);
        conv_scalar_kernel<<<grid1d((long)B * 2 * H * W), dim3(256), 0, stream>>>(
            xd, est_w[l][6], est_b[l][6], fB, nullptr, B, H, W, 640, 0, chin + 448, sgL, 2, 3, 1);

        if (l == 4) {
            // context network: append new flows at channels [628,632)
            flow_to_xd_kernel<<<grid1d((long)B * H * W), dim3(256), 0, stream>>>(fA, fB, xd, B, H, W, 640, 628);
            Seg4 sgC = estSeg(0, true, 8);           // [624,632) <-> ref [610,618)
            int chunks0 = ceilDiv(632, 32);
            unsigned short* wp0 = (unsigned short*)alloc(packBytes(128, chunks0));
            launch_pack(ctx_w[0], wp0, 128, 618, chunks0, sgC, stream);
            launch_conv(xd, wp0, ctx_b[0], cxA, B, H, W, 640, 0, H, W, 128, 128, 0, 1, 1, chunks0, 1, stream);
            int cins[6]  = { 618, 128, 128, 128, 96, 64 };
            int couts[6] = { 128, 128, 128, 96, 64, 32 };
            int dils[6]  = { 1, 2, 4, 8, 16, 1 };
            unsigned short* src = cxA; unsigned short* dst = cxB;
            for (int j = 1; j < 6; ++j) {
                int chunks = ceilDiv(cins[j], 32);
                unsigned short* wp = (unsigned short*)alloc(packBytes(couts[j], chunks));
                launch_pack(ctx_w[j], wp, couts[j], cins[j], chunks, identSeg(cins[j]), stream);
                launch_conv(src, wp, ctx_b[j], dst, B, H, W, 128, 0, H, W, couts[j], 128, 0, 1, dils[j], chunks, 1, stream);
                unsigned short* t = src; src = dst; dst = t;
            }
            // flow = flow + ctx_last(cx)
            conv_scalar_kernel<<<grid1d((long)B * 2 * H * W), dim3(256), 0, stream>>>(
                src, ctx_w[6], ctx_b[6], fA, fA, B, H, W, 128, 0, 32, identSeg(32), 2, 3, 1);
        }
        pHf = H; pWf = W;
    }

    // final: upsample flow to full res * (1/DIV_FLOW)
    upsample_kernel<<<grid1d((long)B * 2 * Him * Wim), dim3(256), 0, stream>>>(
        fA, (float*)d_out, B, 2, 96, 192, Him, Wim, 20.f);
}
